// RPGD_23716809408742
// MI455X (gfx1250) — compile-verified, matrix operand order checked by reading, not run on hardware
//
#include <hip/hip_runtime.h>
#include <hip/hip_fp16.h>
#include <stdint.h>

// ---------------------------------------------------------------------------
// MI455X (gfx1250, wave32) implementation.
//  * ~200 GFLOP total, all operands L2-resident (192MB) -> compute bound.
//  * All hot matmuls lowered to v_wmma_f32_16x16x32_f16 (f16 in, f32 acc),
//    batching 4 items x 4 tokens into the 16-row WMMA tile.
//  * Weights pre-swizzled into exact WMMA B lane layout; encoder tiles cached
//    in LDS (CDNA5: up to 320KB/WG), 512KB bilinear B streams from L2.
// ---------------------------------------------------------------------------

typedef __attribute__((ext_vector_type(16))) _Float16 v16h;
typedef __attribute__((ext_vector_type(8)))  float    v8f;

#define Dm 64
#define MSLOTS 4
#define RTYPES 5

// LDS strides (padded against bank conflicts)
#define XS 68    // X: f32, 16 rows x 64
#define QS 200   // QKV: f16, 16 rows x 192
#define OS 72    // O: f16, 16 rows x 64
#define FS 32    // FA: f16, 16 rows x 32

#define WAVE_BYTES (16*XS*4 + 16*QS*2 + 16*OS*2 + 16*FS*2)   // 14080
#define HDR_BYTES  (2*32*XS*4 + 3*32*4 + 3*32*4)             // 18176
#define ENC_TILES  38
#define ENC_HALVES (ENC_TILES*512)
#define MEMW_TILES 512
#define SMEM_TOTAL (HDR_BYTES + ENC_HALVES*2 + 8*WAVE_BYTES) // 169728

__device__ __forceinline__ v8f wmma16(v16h a, v16h b, v8f c) {
  return __builtin_amdgcn_wmma_f32_16x16x32_f16(false, a, false, b, (short)0, c, false, false);
}

// A operand (16x32 f16): element (M,K): lane = M + 16*((K%16)>=8);
// halves j<8 <-> K = koff+j, j>=8 <-> K = 16+koff+(j-8), koff = 8*(lane>=16).
__device__ __forceinline__ v16h loadA_f32(const float* Xb, int stride, int kbase, int lane) {
  const int mr = lane & 15;
  const int ko = (lane & 16) ? 8 : 0;
  const float* p = Xb + mr * stride + kbase + ko;
  v16h a;
#pragma unroll
  for (int j = 0; j < 8; ++j) a[j] = (_Float16)p[j];
#pragma unroll
  for (int j = 0; j < 8; ++j) a[j + 8] = (_Float16)p[16 + j];
  return a;
}
__device__ __forceinline__ v16h loadA_f16(const _Float16* Xb, int stride, int kbase, int lane) {
  const int mr = lane & 15;
  const int ko = (lane & 16) ? 8 : 0;
  const _Float16* p = Xb + mr * stride + kbase + ko;
  v16h a;
#pragma unroll
  for (int j = 0; j < 8; ++j) a[j] = p[j];
#pragma unroll
  for (int j = 0; j < 8; ++j) a[j + 8] = p[16 + j];
  return a;
}
// B operand tiles are pre-swizzled: 512 halves/tile, lane-major, 32B per lane.
__device__ __forceinline__ v16h loadB(const _Float16* base, int tile, int lane) {
  return *((const v16h*)base + tile * 32 + lane);
}

struct EncP {
  const _Float16* B;   // tiles: [0,24) in_w, [24,32) out_w, [32,34) ff1, [34,38) ff2
  const float *in_b, *out_b, *ff1_b, *ff2_b;
  const float *ln1_g, *ln1_b, *ln2_g, *ln2_b;
};

__device__ __forceinline__ void layer_norm16(float* X, const float* g, const float* b, int lane) {
  const int r = lane & 15;
  const int d0 = (lane & 16) ? 32 : 0;
  float* p = X + r * XS;
  float s = 0.f, sq = 0.f;
#pragma unroll
  for (int d = 0; d < 32; ++d) { const float x = p[d0 + d]; s += x; sq += x * x; }
  s  += __shfl_xor(s, 16, 32);
  sq += __shfl_xor(sq, 16, 32);
  const float mu = s * (1.f / 64.f);
  const float var = sq * (1.f / 64.f) - mu * mu;
  const float rinv = rsqrtf(var + 1e-5f);
#pragma unroll
  for (int d = 0; d < 32; ++d) {
    const int dd = d0 + d;
    p[dd] = (p[dd] - mu) * rinv * g[dd] + b[dd];
  }
}

// One post-norm TransformerEncoderLayer pass on X (16 rows = 4 items x 4 tokens).
__device__ __forceinline__ void encoder_pass(float* X, _Float16* QKV, _Float16* O, _Float16* FA,
                                             const EncP& w, int lane) {
  const int nn = lane & 15;
  const int mb = (lane & 16) ? 8 : 0;
  { // QKV = X @ in_w^T + in_b  (16x64 @ 64x192)
    v16h a0 = loadA_f32(X, XS, 0, lane);
    v16h a1 = loadA_f32(X, XS, 32, lane);
#pragma unroll
    for (int nt = 0; nt < 12; ++nt) {
      v8f c = {};
      c = wmma16(a0, loadB(w.B, nt * 2 + 0, lane), c);
      c = wmma16(a1, loadB(w.B, nt * 2 + 1, lane), c);
      const float bias = w.in_b[nt * 16 + nn];
#pragma unroll
      for (int v = 0; v < 8; ++v) QKV[(mb + v) * QS + nt * 16 + nn] = (_Float16)(c[v] + bias);
    }
  }
  __syncthreads();
  { // attention: seq 4, heads 4x16; each lane: one row, two heads
    const int r = lane & 15;
    const int bq = r >> 2;
    const int h0 = (lane >> 4) * 2;
#pragma unroll
    for (int hh = 0; hh < 2; ++hh) {
      const int qc = (h0 + hh) * 16;
      float q[16];
#pragma unroll
      for (int k = 0; k < 16; ++k) q[k] = (float)QKV[r * QS + qc + k];
      float s[4];
#pragma unroll
      for (int kt = 0; kt < 4; ++kt) {
        const _Float16* kp = QKV + ((bq << 2) + kt) * QS + 64 + qc;
        float acc = 0.f;
#pragma unroll
        for (int k = 0; k < 16; ++k) acc += q[k] * (float)kp[k];
        s[kt] = acc * 0.25f;                       // 1/sqrt(16)
      }
      const float mx = fmaxf(fmaxf(s[0], s[1]), fmaxf(s[2], s[3]));
      float p[4]; float sum = 0.f;
#pragma unroll
      for (int kt = 0; kt < 4; ++kt) { p[kt] = __expf(s[kt] - mx); sum += p[kt]; }
      const float ri = 1.f / sum;
#pragma unroll
      for (int j = 0; j < 16; ++j) {
        float o = 0.f;
#pragma unroll
        for (int kt = 0; kt < 4; ++kt) o += p[kt] * (float)QKV[((bq << 2) + kt) * QS + 128 + qc + j];
        O[r * OS + qc + j] = (_Float16)(o * ri);
      }
    }
  }
  __syncthreads();
  { // out projection + residual
    v16h a0 = loadA_f16(O, OS, 0, lane);
    v16h a1 = loadA_f16(O, OS, 32, lane);
#pragma unroll
    for (int nt = 0; nt < 4; ++nt) {
      v8f c = {};
      c = wmma16(a0, loadB(w.B, 24 + nt * 2 + 0, lane), c);
      c = wmma16(a1, loadB(w.B, 24 + nt * 2 + 1, lane), c);
      const float bias = w.out_b[nt * 16 + nn];
#pragma unroll
      for (int v = 0; v < 8; ++v) X[(mb + v) * XS + nt * 16 + nn] += c[v] + bias;
    }
  }
  __syncthreads();
  layer_norm16(X, w.ln1_g, w.ln1_b, lane);
  __syncthreads();
  { // FF1 + ReLU (output 16 cols; cols 16..31 zero-padded for the K=32 FF2 tile)
    v16h a0 = loadA_f32(X, XS, 0, lane);
    v16h a1 = loadA_f32(X, XS, 32, lane);
    v8f c = {};
    c = wmma16(a0, loadB(w.B, 32, lane), c);
    c = wmma16(a1, loadB(w.B, 33, lane), c);
    const float bias = w.ff1_b[nn];
#pragma unroll
    for (int v = 0; v < 8; ++v) {
      FA[(mb + v) * FS + nn]      = (_Float16)fmaxf(c[v] + bias, 0.f);
      FA[(mb + v) * FS + 16 + nn] = (_Float16)0.f;
    }
  }
  __syncthreads();
  { // FF2 + residual
    v16h a = loadA_f16(FA, FS, 0, lane);
#pragma unroll
    for (int nt = 0; nt < 4; ++nt) {
      v8f c = {};
      c = wmma16(a, loadB(w.B, 34 + nt, lane), c);
      const float bias = w.ff2_b[nt * 16 + nn];
#pragma unroll
      for (int v = 0; v < 8; ++v) X[(mb + v) * XS + nt * 16 + nn] += c[v] + bias;
    }
  }
  __syncthreads();
  layer_norm16(X, w.ln2_g, w.ln2_b, lane);
  __syncthreads();
}

// ---------------------------------------------------------------------------
// Prep: swizzle all weights (f32 -> f16) into WMMA B lane layout.
// tile t, thread q=0..511: lane = q>>4, j = q&15.
extern "C" __global__ void rpgd_prep_weights(const float* __restrict__ in_w,
                                             const float* __restrict__ out_w,
                                             const float* __restrict__ ff1_w,
                                             const float* __restrict__ ff2_w,
                                             const float* __restrict__ memw_w,
                                             _Float16* __restrict__ wsB_enc,
                                             _Float16* __restrict__ wsB_memw) {
  const int t = blockIdx.x;
  const int q = threadIdx.x;
  const int ln = q >> 4, j = q & 15;
  const int nn = ln & 15;
  const int hi = ln >> 4;
  const int kk = (j < 8) ? (hi * 8 + j) : (16 + hi * 8 + (j - 8));
  float val;
  _Float16* dp;
  if (t < MEMW_TILES) { // bilinear B: B[K=i*64+d][o] = memw_w[o*64+i, d]
    const int kc = t >> 2, nt = t & 3;
    const int K = kc * 32 + kk;
    const int i = K >> 6, d = K & 63;
    const int o = nt * 16 + nn;
    val = memw_w[(size_t)(o * 64 + i) * 64 + d];
    dp = wsB_memw + (size_t)t * 512 + q;
  } else {
    const int u = t - MEMW_TILES;
    if (u < 24)      { const int nt = u >> 1, ck = u & 1;  val = in_w[(nt * 16 + nn) * 64 + ck * 32 + kk]; }
    else if (u < 32) { const int v2 = u - 24; const int nt = v2 >> 1, ck = v2 & 1;
                       val = out_w[(nt * 16 + nn) * 64 + ck * 32 + kk]; }
    else if (u < 34) { const int ck = u - 32;              val = ff1_w[nn * 64 + ck * 32 + kk]; }
    else             { const int nt = u - 34;              val = (kk < 16) ? ff2_w[(nt * 16 + nn) * 16 + kk] : 0.f; }
    dp = wsB_enc + (size_t)u * 512 + q;
  }
  *dp = (_Float16)val;
}

// ---------------------------------------------------------------------------
// Stable rank of each edge within its etype (original order), + counts/start.
extern "C" __global__ void __launch_bounds__(1024)
rpgd_etype_scan(const int* __restrict__ etype, int E,
                int* __restrict__ elocal, int* __restrict__ counts, int* __restrict__ startv) {
  __shared__ int run[RTYPES];
  __shared__ int wcnt[32][RTYPES];
  __shared__ int woff[32][RTYPES];
  const int tid = threadIdx.x;
  const int lane = tid & 31;
  const int wv = tid >> 5;
  if (tid < RTYPES) run[tid] = 0;
  __syncthreads();
  for (int base = 0; base < E; base += 1024) {
    const int e = base + tid;
    const int et = (e < E) ? etype[e] : -1;
    int rank = 0;
#pragma unroll
    for (int r = 0; r < RTYPES; ++r) {
      const unsigned long long b = __ballot(et == r);
      if (lane == 0) wcnt[wv][r] = (int)__popcll(b);
      if (et == r) rank = (int)__popcll(b & ((1ull << lane) - 1ull));
    }
    __syncthreads();
    if (tid == 0) {
      for (int r = 0; r < RTYPES; ++r) {
        int a = run[r];
        for (int w2 = 0; w2 < 32; ++w2) { woff[w2][r] = a; a += wcnt[w2][r]; }
        run[r] = a;
      }
    }
    __syncthreads();
    if (e < E) elocal[e] = woff[wv][et] + rank;
    __syncthreads();
  }
  if (tid == 0) {
    int a = 0;
    for (int r = 0; r < RTYPES; ++r) { counts[r] = run[r]; startv[r] = a; a += run[r]; }
  }
}

extern "C" __global__ void rpgd_zero(float* __restrict__ p, long long n) {
  long long i = (long long)blockIdx.x * blockDim.x + threadIdx.x;
  const long long st = (long long)gridDim.x * blockDim.x;
  for (; i < n; i += st) p[i] = 0.f;
}

extern "C" __global__ void rpgd_deg(const int* __restrict__ dst, float* __restrict__ deg, int E) {
  const int e = blockIdx.x * blockDim.x + threadIdx.x;
  if (e < E) __hip_atomic_fetch_add(&deg[dst[e]], 1.f, __ATOMIC_RELAXED, __HIP_MEMORY_SCOPE_AGENT);
}

// ---------------------------------------------------------------------------
// Edge kernel: 256 threads = 8 waves, 32 edges/WG (4 edges/wave in the encoder
// phase; 2 Mtile x 4 Ntile split of the K=4096 bilinear in the GEMM phase).
// Requires E % 32 == 0 (holds: E = 160000).
extern "C" __global__ void __launch_bounds__(256)
rpgd_edge_kernel(const float* __restrict__ feat, const int* __restrict__ src,
                 const int* __restrict__ dst, const int* __restrict__ etype,
                 const float* __restrict__ cls_w, const float* __restrict__ sep,
                 const float* __restrict__ in_b, const float* __restrict__ out_b,
                 const float* __restrict__ ln1_g, const float* __restrict__ ln1_b,
                 const float* __restrict__ ln2_g, const float* __restrict__ ln2_b,
                 const float* __restrict__ ff1_b, const float* __restrict__ ff2_b,
                 const float* __restrict__ memcat_w, const float* __restrict__ memcat_b,
                 const _Float16* __restrict__ wsB_enc, const _Float16* __restrict__ wsB_memw,
                 const int* __restrict__ elocal, const int* __restrict__ counts,
                 const int* __restrict__ startv,
                 float* __restrict__ agg, float* __restrict__ category, int* __restrict__ label) {
  extern __shared__ unsigned char smem[];
  float* hsrc  = (float*)smem;                 // 32 x XS
  float* memem = hsrc + 32 * XS;               // 32 x XS
  int* dstl = (int*)(memem + 32 * XS);
  int* etl  = dstl + 32;
  int* locl = etl + 32;
  float* statp = (float*)(locl + 32);          // 96 floats (used by node kernel)
  _Float16* encB = (_Float16*)(statp + 96);
  unsigned char* pw = (unsigned char*)(encB + ENC_HALVES);

  const int tid = threadIdx.x;
  const int lane = tid & 31;
  const int wave = tid >> 5;
  float* X = (float*)(pw + (size_t)wave * WAVE_BYTES);
  _Float16* QKV = (_Float16*)(X + 16 * XS);
  _Float16* O = QKV + 16 * QS;
  _Float16* FA = O + 16 * OS;

  const int e0 = blockIdx.x * 32;

  for (int i = tid; i < ENC_HALVES; i += 256) encB[i] = wsB_enc[i];
  for (int i = tid; i < 32 * Dm; i += 256) {
    const int e = i >> 6, d = i & 63;
    hsrc[e * XS + d] = feat[(size_t)src[e0 + e] * Dm + d];
    memem[e * XS + d] = 0.f;
  }
  if (tid < 32) {
    dstl[tid] = dst[e0 + tid];
    etl[tid]  = etype[e0 + tid];
    locl[tid] = elocal[e0 + tid];
  }
  __syncthreads();

  EncP w;
  w.B = encB; w.in_b = in_b; w.out_b = out_b; w.ff1_b = ff1_b; w.ff2_b = ff2_b;
  w.ln1_g = ln1_g; w.ln1_b = ln1_b; w.ln2_g = ln2_g; w.ln2_b = ln2_b;

  const int eb = wave * 4;
  for (int m = 0; m < MSLOTS; ++m) {
    { // build sequence rows [CLS_m, h_src, SEP, h_dst] per edge
      const int r = lane & 15;
      const int d0 = (lane & 16) ? 32 : 0;
      const int b = r >> 2, t = r & 3;
      const float* sp;
      if (t == 0)      sp = cls_w + m * Dm + d0;
      else if (t == 1) sp = hsrc + (eb + b) * XS + d0;
      else if (t == 2) sp = sep + d0;
      else             sp = feat + (size_t)dstl[eb + b] * Dm + d0;
#pragma unroll
      for (int d = 0; d < 32; ++d) X[r * XS + d0 + d] = sp[d];
    }
    __syncthreads();
    encoder_pass(X, QKV, O, FA, w, lane);
    if (lane < 16) { // category + label (etype-ordered scatter is a bijection)
      const int b = lane >> 2, j = lane & 3;
      const float* xr = X + (b << 2) * XS;
      float acc = memcat_b[j];
      for (int d = 0; d < Dm; ++d) acc += xr[d] * memcat_w[j * Dm + d];
      const int el = eb + b;
      const int et = etl[el];
      const int pos = MSLOTS * startv[et] + m * counts[et] + locl[el];
      category[(size_t)pos * MSLOTS + j] = acc;
      if (j == 0) label[pos] = m;
    }
    { // mem_emb += CLS-row output
      const int b = lane >> 3, db = (lane & 7) * 8;
      const float* xr = X + (b << 2) * XS + db;
      float* mp = memem + (eb + b) * XS + db;
#pragma unroll
      for (int d = 0; d < 8; ++d) mp[d] += xr[d];
    }
    __syncthreads();
  }

  // Bilinear GEMM: msg[e,o] = sum_{i,d} memw[o,i,d]*memem[e,d]*hsrc[e,i]
  // A built on the fly as P[e, i*64+d] = hsrc[e,i]*memem[e,d]; K = 4096.
  const int Mt = wave & 1, Nt = wave >> 1;
  const int mr = lane & 15;
  const int ko = (lane & 16) ? 8 : 0;
  const float* hs = hsrc + (Mt * 16 + mr) * XS;
  const float* me = memem + (Mt * 16 + mr) * XS;
  v8f acc = {};
  for (int kc = 0; kc < 128; ++kc) {
    const int i = kc >> 1;
    const int db = (kc & 1) * 32;
    const float hv = hs[i];
    v16h a;
#pragma unroll
    for (int j = 0; j < 8; ++j) a[j]     = (_Float16)(hv * me[db + ko + j]);
#pragma unroll
    for (int j = 0; j < 8; ++j) a[j + 8] = (_Float16)(hv * me[db + 16 + ko + j]);
    acc = wmma16(a, loadB(wsB_memw, kc * 4 + Nt, lane), acc);
  }
  const int nn = lane & 15;
  const int mb = (lane & 16) ? 8 : 0;
#pragma unroll
  for (int v = 0; v < 8; ++v) {
    const int el = Mt * 16 + mb + v;
    const int o = Nt * 16 + nn;
    __hip_atomic_fetch_add(&agg[(size_t)dstl[el] * Dm + o], acc[v],
                           __ATOMIC_RELAXED, __HIP_MEMORY_SCOPE_AGENT);
  }
}

// ---------------------------------------------------------------------------
// Node kernel: same encoder with (feat, feat); fuses mean-agg LN + h_bias +
// self bilinear + LeakyReLU(0.2). Requires N % 32 == 0 (N = 20000).
extern "C" __global__ void __launch_bounds__(256)
rpgd_node_kernel(const float* __restrict__ feat,
                 const float* __restrict__ cls_w, const float* __restrict__ sep,
                 const float* __restrict__ in_b, const float* __restrict__ out_b,
                 const float* __restrict__ ln1_g, const float* __restrict__ ln1_b,
                 const float* __restrict__ ln2_g, const float* __restrict__ ln2_b,
                 const float* __restrict__ ff1_b, const float* __restrict__ ff2_b,
                 const _Float16* __restrict__ wsB_enc, const _Float16* __restrict__ wsB_memw,
                 const float* __restrict__ agg, const float* __restrict__ deg,
                 const float* __restrict__ lnw_g, const float* __restrict__ lnw_b,
                 const float* __restrict__ h_bias,
                 float* __restrict__ out) {
  extern __shared__ unsigned char smem[];
  float* hsrc  = (float*)smem;
  float* memem = hsrc + 32 * XS;
  int* dstl = (int*)(memem + 32 * XS);     // unused, keeps layout identical
  float* statp = (float*)(dstl + 96);
  float* amu = statp;
  float* arv = statp + 32;
  float* adg = statp + 64;
  _Float16* encB = (_Float16*)(statp + 96);
  unsigned char* pw = (unsigned char*)(encB + ENC_HALVES);

  const int tid = threadIdx.x;
  const int lane = tid & 31;
  const int wave = tid >> 5;
  float* X = (float*)(pw + (size_t)wave * WAVE_BYTES);
  _Float16* QKV = (_Float16*)(X + 16 * XS);
  _Float16* O = QKV + 16 * QS;
  _Float16* FA = O + 16 * OS;

  const int n0 = blockIdx.x * 32;

  for (int i = tid; i < ENC_HALVES; i += 256) encB[i] = wsB_enc[i];
  for (int i = tid; i < 32 * Dm; i += 256) {
    const int e = i >> 6, d = i & 63;
    hsrc[e * XS + d] = feat[(size_t)(n0 + e) * Dm + d];
    memem[e * XS + d] = 0.f;
  }
  if (tid < 32) { // LN stats of mean-aggregated messages
    const int n = n0 + tid;
    const float di = 1.f / fmaxf(deg[n], 1.f);
    float s = 0.f, sq = 0.f;
    const float* ap = agg + (size_t)n * Dm;
    for (int d = 0; d < Dm; ++d) { const float a = ap[d] * di; s += a; sq += a * a; }
    const float mu = s * (1.f / 64.f);
    const float var = sq * (1.f / 64.f) - mu * mu;
    amu[tid] = mu; arv[tid] = rsqrtf(var + 1e-5f); adg[tid] = di;
  }
  __syncthreads();

  EncP w;
  w.B = encB; w.in_b = in_b; w.out_b = out_b; w.ff1_b = ff1_b; w.ff2_b = ff2_b;
  w.ln1_g = ln1_g; w.ln1_b = ln1_b; w.ln2_g = ln2_g; w.ln2_b = ln2_b;

  const int eb = wave * 4;
  for (int m = 0; m < MSLOTS; ++m) {
    {
      const int r = lane & 15;
      const int d0 = (lane & 16) ? 32 : 0;
      const int b = r >> 2, t = r & 3;
      const float* sp;
      if (t == 0)      sp = cls_w + m * Dm + d0;
      else if (t == 2) sp = sep + d0;
      else             sp = hsrc + (eb + b) * XS + d0;   // tokens 1 and 3 = feat[n]
#pragma unroll
      for (int d = 0; d < 32; ++d) X[r * XS + d0 + d] = sp[d];
    }
    __syncthreads();
    encoder_pass(X, QKV, O, FA, w, lane);
    {
      const int b = lane >> 3, db = (lane & 7) * 8;
      const float* xr = X + (b << 2) * XS + db;
      float* mp = memem + (eb + b) * XS + db;
#pragma unroll
      for (int d = 0; d < 8; ++d) mp[d] += xr[d];
    }
    __syncthreads();
  }

  const int Mt = wave & 1, Nt = wave >> 1;
  const int mr = lane & 15;
  const int ko = (lane & 16) ? 8 : 0;
  const float* hs = hsrc + (Mt * 16 + mr) * XS;
  const float* me = memem + (Mt * 16 + mr) * XS;
  v8f acc = {};
  for (int kc = 0; kc < 128; ++kc) {
    const int i = kc >> 1;
    const int db = (kc & 1) * 32;
    const float hv = hs[i];
    v16h a;
#pragma unroll
    for (int j = 0; j < 8; ++j) a[j]     = (_Float16)(hv * me[db + ko + j]);
#pragma unroll
    for (int j = 0; j < 8; ++j) a[j + 8] = (_Float16)(hv * me[db + 16 + ko + j]);
    acc = wmma16(a, loadB(wsB_memw, kc * 4 + Nt, lane), acc);
  }
  const int nn = lane & 15;
  const int mb = (lane & 16) ? 8 : 0;
#pragma unroll
  for (int v = 0; v < 8; ++v) {
    const int nl = Mt * 16 + mb + v;
    const int n = n0 + nl;
    const int o = Nt * 16 + nn;
    float val = (agg[(size_t)n * Dm + o] * adg[nl] - amu[nl]) * arv[nl] * lnw_g[o]
                + lnw_b[o] + h_bias[o] + acc[v];
    out[(size_t)n * Dm + o] = (val > 0.f) ? val : 0.2f * val;
  }
}

// ---------------------------------------------------------------------------
extern "C" void kernel_launch(void* const* d_in, const int* in_sizes, int n_in,
                              void* d_out, int out_size, void* d_ws, size_t ws_size,
                              hipStream_t stream) {
  const float* feat   = (const float*)d_in[0];
  const int*   src    = (const int*)d_in[1];
  const int*   dst    = (const int*)d_in[2];
  const int*   etype  = (const int*)d_in[3];
  const float* cls_w  = (const float*)d_in[4];
  const float* sep    = (const float*)d_in[5];
  const float* in_w   = (const float*)d_in[6];
  const float* in_b   = (const float*)d_in[7];
  const float* out_w  = (const float*)d_in[8];
  const float* out_b  = (const float*)d_in[9];
  const float* ln1_g  = (const float*)d_in[10];
  const float* ln1_b  = (const float*)d_in[11];
  const float* ln2_g  = (const float*)d_in[12];
  const float* ln2_b  = (const float*)d_in[13];
  const float* ff1_w  = (const float*)d_in[14];
  const float* ff1_b  = (const float*)d_in[15];
  const float* ff2_w  = (const float*)d_in[16];
  const float* ff2_b  = (const float*)d_in[17];
  const float* memcat_w = (const float*)d_in[18];
  const float* memcat_b = (const float*)d_in[19];
  const float* memw_w   = (const float*)d_in[20];
  const float* h_bias   = (const float*)d_in[21];
  const float* lnw_g    = (const float*)d_in[22];
  const float* lnw_b    = (const float*)d_in[23];

  const int N = in_sizes[0] / Dm;
  const int E = in_sizes[1];

  // workspace carve (~6.4 MB)
  float* agg = (float*)d_ws;
  float* deg = agg + (size_t)N * Dm;
  int* elocal = (int*)(deg + N);
  int* counts = elocal + E;
  int* startv = counts + 8;
  uintptr_t hb = (uintptr_t)(startv + 8);
  hb = (hb + 255) & ~(uintptr_t)255;
  _Float16* wsB_enc  = (_Float16*)hb;
  _Float16* wsB_memw = wsB_enc + ENC_HALVES;

  rpgd_prep_weights<<<MEMW_TILES + ENC_TILES, 512, 0, stream>>>(
      in_w, out_w, ff1_w, ff2_w, memw_w, wsB_enc, wsB_memw);
  rpgd_etype_scan<<<1, 1024, 0, stream>>>(etype, E, elocal, counts, startv);
  rpgd_zero<<<512, 256, 0, stream>>>(agg, (long long)N * Dm + N);
  rpgd_deg<<<(E + 255) / 256, 256, 0, stream>>>(dst, deg, E);

  float* out_node = (float*)d_out;
  float* out_cat  = out_node + (size_t)N * Dm;
  int*   out_label = (int*)(out_cat + (size_t)MSLOTS * E * MSLOTS);

  (void)hipFuncSetAttribute((const void*)rpgd_edge_kernel,
                            hipFuncAttributeMaxDynamicSharedMemorySize, SMEM_TOTAL);
  (void)hipFuncSetAttribute((const void*)rpgd_node_kernel,
                            hipFuncAttributeMaxDynamicSharedMemorySize, SMEM_TOTAL);

  rpgd_edge_kernel<<<E / 32, 256, SMEM_TOTAL, stream>>>(
      feat, src, dst, etype, cls_w, sep, in_b, out_b, ln1_g, ln1_b, ln2_g, ln2_b,
      ff1_b, ff2_b, memcat_w, memcat_b, wsB_enc, wsB_memw, elocal, counts, startv,
      agg, out_cat, out_label);

  rpgd_node_kernel<<<N / 32, 256, SMEM_TOTAL, stream>>>(
      feat, cls_w, sep, in_b, out_b, ln1_g, ln1_b, ln2_g, ln2_b, ff1_b, ff2_b,
      wsB_enc, wsB_memw, agg, deg, lnw_g, lnw_b, h_bias, out_node);
}
